// DiagonalWindowAttention_34797825032706
// MI455X (gfx1250) — matrix operand order repaired and checked by reading, not compiled, not run on hardware
//
#include <hip/hip_runtime.h>
#include <hip/hip_bf16.h>

// ---------------- static config (matches reference) ----------------
#define NWIN   1024
#define LTOK   256
#define EMBED  192
#define NHEAD  6
#define HD     32
#define NP     64      // patches per window
#define S2     4       // shuffle groups
#define KDIM   128     // S2*HD, attention contraction dim per head
#define NWM    64      // mask windows
#define SCALE  0.08838834764831845f   // 32^-0.5 / 2
#define EPS    1e-5f

typedef __attribute__((ext_vector_type(16))) __bf16 v16bf;
typedef __attribute__((ext_vector_type(8)))  float  v8f;
typedef int v4i __attribute__((vector_size(16)));   // matches async-LDS builtin param

union FragBF { unsigned int u[8]; v16bf v; };

#define AS1 __attribute__((address_space(1)))
#define AS3 __attribute__((address_space(3)))

#if defined(__gfx1250__) && __has_builtin(__builtin_amdgcn_global_load_async_to_lds_b128)
#define HAVE_ASYNC_LDS 1
#else
#define HAVE_ASYNC_LDS 0
#endif

#if HAVE_ASYNC_LDS
__device__ __forceinline__ void async_b128(const void* g, void* l) {
    __builtin_amdgcn_global_load_async_to_lds_b128(
        (AS1 v4i*)(AS1 void*)(void*)g,
        (AS3 v4i*)(AS3 void*)(void*)l, 0, 0);
}
#endif

__device__ __forceinline__ void wait_async_zero() {
#if HAVE_ASYNC_LDS
#if __has_builtin(__builtin_amdgcn_s_wait_asynccnt)
    __builtin_amdgcn_s_wait_asynccnt(0);
#else
    asm volatile("s_wait_asynccnt 0" ::: "memory");
#endif
#endif
}

__device__ __forceinline__ unsigned short f2bf(float f) {
    unsigned int u = __float_as_uint(f);
    u += 0x7FFFu + ((u >> 16) & 1u);      // round-to-nearest-even
    return (unsigned short)(u >> 16);
}

// ================================================================
// Kernel 1: group-shuffle + LayerNorm(s2*embed) + head split, f32 -> bf16
// out layout: (nw, nh, p, g*32+hd)  [head-major so attention reads are dense]
// ================================================================
__global__ __launch_bounds__(256) void k_ln_pack(
    const float* __restrict__ q, const float* __restrict__ k, const float* __restrict__ vv,
    const float* __restrict__ wq, const float* __restrict__ bq,
    const float* __restrict__ wk, const float* __restrict__ bk,
    const float* __restrict__ wv, const float* __restrict__ bv,
    unsigned short* __restrict__ oq, unsigned short* __restrict__ okk,
    unsigned short* __restrict__ ov)
{
    const int nw = blockIdx.x;
    const int tsel = blockIdx.y;                       // 0:q 1:k 2:v
    const float* x = tsel == 0 ? q  : (tsel == 1 ? k  : vv);
    const float* w = tsel == 0 ? wq : (tsel == 1 ? wk : wv);
    const float* b = tsel == 0 ? bq : (tsel == 1 ? bk : bv);
    unsigned short* o = tsel == 0 ? oq : (tsel == 1 ? okk : ov);
    const float scale = tsel == 0 ? SCALE : 1.0f;

    const int wave = threadIdx.x >> 5;
    const int lane = threadIdx.x & 31;
    const size_t ibase = (size_t)nw * LTOK * EMBED;

    for (int pi = 0; pi < 8; ++pi) {
        const int p = wave * 8 + pi;                   // patch 0..63
        float vals[24];
        float s = 0.f, s2 = 0.f;
        #pragma unroll
        for (int i = 0; i < 24; ++i) {                 // 768 values / 32 lanes
            const int j = i * 32 + lane;               // (g,e) flat
            const int g = j / EMBED, e = j - g * EMBED;
            const int l = ((p >> 3) << 5) + ((g >> 1) << 4) + ((p & 7) << 1) + (g & 1);
            const float xv = x[ibase + (size_t)l * EMBED + e];
            vals[i] = xv; s += xv; s2 += xv * xv;
        }
        #pragma unroll
        for (int m = 16; m >= 1; m >>= 1) { s += __shfl_xor(s, m); s2 += __shfl_xor(s2, m); }
        const float mu  = s * (1.f / 768.f);
        const float var = s2 * (1.f / 768.f) - mu * mu;
        const float rsq = rsqrtf(var + EPS);
        #pragma unroll
        for (int i = 0; i < 24; ++i) {
            const int j = i * 32 + lane;
            const int g = j / EMBED, e = j - g * EMBED;
            const float y = ((vals[i] - mu) * rsq * w[j] + b[j]) * scale;
            const int nh = e >> 5, hd = e & 31;
            o[(((size_t)nw * NHEAD + nh) * NP + p) * KDIM + (g << 5) + hd] = f2bf(y);
        }
    }
}

// ================================================================
// Kernel 2: fused attention per (window, head).
// 4 waves; wave w owns rows [16w,16w+16) of the 64x64 score matrix.
// Q/K staged with GLOBAL_LOAD_ASYNC_TO_LDS_B128 when available.
// Pitches: rows 16B-aligned for B128 DMA and bank-conflict-free for the
// strided fragment reads (68 dw stride -> 4*lane mod 64, distinct).
// ================================================================
#define QP 136   // Q/K LDS row pitch (ushort), 272 B: 16B-aligned
#define VP 66    // V^T LDS row pitch
#define AP 66    // attn-prob LDS row pitch

__global__ __launch_bounds__(128) void k_attn(
    const unsigned short* __restrict__ Q, const unsigned short* __restrict__ K,
    const unsigned short* __restrict__ V,
    const float* __restrict__ bias_table, const float* __restrict__ mask,
    unsigned short* __restrict__ X2)
{
    __shared__ __attribute__((aligned(16))) unsigned short qlds[64 * QP];
    __shared__ __attribute__((aligned(16))) unsigned short klds[64 * QP];
    __shared__ __attribute__((aligned(16))) unsigned short vlds[128 * VP];  // V^T: [j][p]
    __shared__ __attribute__((aligned(16))) unsigned short alds[64 * AP];   // probs

    const int bid = blockIdx.x;
    const int nw = bid / NHEAD, nh = bid - nw * NHEAD;
    const size_t base = ((size_t)nw * NHEAD + nh) * NP * KDIM;
    const int t = threadIdx.x;

#if HAVE_ASYNC_LDS
    // async DMA: 64 rows x 16 chunks of 16B per tensor, no VGPR round-trip
    for (int idx = t; idx < 1024; idx += 128) {
        const int p = idx >> 4, c8 = (idx & 15) << 3;
        async_b128(Q + base + (size_t)p * KDIM + c8, &qlds[p * QP + c8]);
        async_b128(K + base + (size_t)p * KDIM + c8, &klds[p * QP + c8]);
    }
#else
    for (int idx = t; idx < 4096; idx += 128) {
        const int p = idx >> 6, c2 = (idx & 63) << 1;
        *(unsigned int*)&qlds[p * QP + c2] = *(const unsigned int*)&Q[base + (size_t)p * KDIM + c2];
        *(unsigned int*)&klds[p * QP + c2] = *(const unsigned int*)&K[base + (size_t)p * KDIM + c2];
    }
#endif
    // V transposed (element-wise; DMA cannot transpose 16-bit data)
    for (int idx = t; idx < 8192; idx += 128) {
        const int p = idx >> 7, j = idx & 127;
        vlds[j * VP + p] = V[base + (size_t)p * KDIM + j];
    }
    wait_async_zero();
    __syncthreads();

    const int wid = t >> 5, lane = t & 31;
    const int mbase = wid * 16;
    const int lh = lane >> 4, ll = lane & 15;

    // ---- scores = Q K^T  (64x64, K=128 via 4 chained bf16 WMMAs) ----
    v8f acc[4] = { {0,0,0,0,0,0,0,0}, {0,0,0,0,0,0,0,0},
                   {0,0,0,0,0,0,0,0}, {0,0,0,0,0,0,0,0} };
    const int arow = mbase + ll;
    for (int ks = 0; ks < 4; ++ks) {
        FragBF a;
        #pragma unroll
        for (int vv = 0; vv < 8; ++vv) {
            const int kk = ks * 32 + ((vv >> 2) << 4) + (lh << 3) + ((vv & 3) << 1);
            a.u[vv] = *(const unsigned int*)&qlds[arow * QP + kk];
        }
        #pragma unroll
        for (int n = 0; n < 4; ++n) {
            FragBF bb;
            const int brow = n * 16 + ll;
            #pragma unroll
            for (int vv = 0; vv < 8; ++vv) {
                const int kk = ks * 32 + ((vv >> 2) << 4) + (lh << 3) + ((vv & 3) << 1);
                bb.u[vv] = *(const unsigned int*)&klds[brow * QP + kk];
            }
            acc[n] = __builtin_amdgcn_wmma_f32_16x16x32_bf16(
                false, a.v, false, bb.v, (short)0, acc[n], false, false);
        }
    }

    // ---- + rel-pos bias + mask, softmax over the 64 columns ----
    const float* mptr = mask + (size_t)(nw & (NWM - 1)) * NP * NP;
    #pragma unroll
    for (int r = 0; r < 8; ++r) {
        const int p = mbase + (lh << 3) + r;
        const int ph = p >> 3, pw = p & 7;
        float e4[4];
        float mx = -1e30f;
        #pragma unroll
        for (int n = 0; n < 4; ++n) {
            const int qc = n * 16 + ll;
            const int ridx = (ph - (qc >> 3) + 7) * 15 + (pw - (qc & 7) + 7);
            const float val = acc[n][r] + bias_table[ridx * NHEAD + nh] + mptr[p * NP + qc];
            e4[n] = val;
            mx = fmaxf(mx, val);
        }
        #pragma unroll
        for (int m = 8; m >= 1; m >>= 1) mx = fmaxf(mx, __shfl_xor(mx, m));
        float sum = 0.f;
        #pragma unroll
        for (int n = 0; n < 4; ++n) { e4[n] = __expf(e4[n] - mx); sum += e4[n]; }
        #pragma unroll
        for (int m = 8; m >= 1; m >>= 1) sum += __shfl_xor(sum, m);
        const float inv = 1.0f / sum;
        #pragma unroll
        for (int n = 0; n < 4; ++n)
            alds[p * AP + n * 16 + ll] = f2bf(e4[n] * inv);   // own rows only
    }
    // no barrier needed: each wave re-reads only its own attn rows

    // ---- out = attn @ V  (16x128 strip per wave, K=64 via 2 WMMAs) ----
    const size_t obase = (size_t)nw * LTOK * EMBED;
    for (int jt = 0; jt < 8; ++jt) {
        v8f o = {0,0,0,0,0,0,0,0};
        #pragma unroll
        for (int ks = 0; ks < 2; ++ks) {
            FragBF a, bb;
            #pragma unroll
            for (int vv = 0; vv < 8; ++vv) {
                const int kk = ks * 32 + ((vv >> 2) << 4) + (lh << 3) + ((vv & 3) << 1);
                a.u[vv] = *(const unsigned int*)&alds[arow * AP + kk];
            }
            const int brow = jt * 16 + ll;                    // output column j
            #pragma unroll
            for (int vv = 0; vv < 8; ++vv) {
                const int kk = ks * 32 + ((vv >> 2) << 4) + (lh << 3) + ((vv & 3) << 1);
                bb.u[vv] = *(const unsigned int*)&vlds[brow * VP + kk];
            }
            o = __builtin_amdgcn_wmma_f32_16x16x32_bf16(
                false, a.v, false, bb.v, (short)0, o, false, false);
        }
        #pragma unroll
        for (int r = 0; r < 8; ++r) {
            const int p = mbase + (lh << 3) + r;
            const int j = jt * 16 + ll;
            const int g = j >> 5, hd = j & 31;
            const int l = ((p >> 3) << 5) + ((g >> 1) << 4) + ((p & 7) << 1) + (g & 1);
            X2[obase + (size_t)l * EMBED + nh * HD + hd] = f2bf(o[r]);   // ungrouped
        }
    }
}

// ================================================================
// Kernel 3: output projection Y = X @ W^T  (262144 x 192) x (192 x 192)
// 64-row tiles; X staged via async DMA; W staged to LDS (f32->bf16) in
// two 96-col passes (keeps LDS <= 63 KB).
// ================================================================
#define XP 200   // 400 B rows: 16B-aligned, 100 dw stride (conflict-free)
#define WP 194

__global__ __launch_bounds__(128) void k_proj(
    const unsigned short* __restrict__ X, const float* __restrict__ W,
    float* __restrict__ Y)
{
    __shared__ __attribute__((aligned(16))) unsigned short xlds[64 * XP];
    __shared__ __attribute__((aligned(16))) unsigned short wlds[96 * WP];

    const int blk = blockIdx.x;
    const int t = threadIdx.x, wid = t >> 5, lane = t & 31;
    const int lh = lane >> 4, ll = lane & 15;
    const size_t xbase = (size_t)blk * 64 * EMBED;

#if HAVE_ASYNC_LDS
    for (int idx = t; idx < 1536; idx += 128) {            // 64 rows x 24 x 16B
        const int row = idx / 24, c8 = (idx - row * 24) << 3;
        async_b128(X + xbase + (size_t)row * EMBED + c8, &xlds[row * XP + c8]);
    }
    wait_async_zero();
#else
    for (int idx = t; idx < 6144; idx += 128) {            // 64x192 bf16 as uints
        const int row = idx / 96, c2 = (idx - row * 96) << 1;
        *(unsigned int*)&xlds[row * XP + c2] =
            *(const unsigned int*)&X[xbase + (size_t)row * EMBED + c2];
    }
#endif

    for (int pass = 0; pass < 2; ++pass) {
        __syncthreads();                                   // pass>0: readers done
        for (int idx = t; idx < 96 * 192; idx += 128) {    // W rows [96p, 96p+96)
            const int o = idx / 192, i = idx - o * 192;
            wlds[o * WP + i] = f2bf(W[(size_t)(pass * 96 + o) * EMBED + i]);
        }
        __syncthreads();

        for (int nt = 0; nt < 6; ++nt) {
            v8f acc = {0,0,0,0,0,0,0,0};
            const int arow = wid * 16 + ll;
            for (int ks = 0; ks < 6; ++ks) {               // K = 192
                FragBF a, bb;
                #pragma unroll
                for (int vv = 0; vv < 8; ++vv) {
                    const int kk = ks * 32 + ((vv >> 2) << 4) + (lh << 3) + ((vv & 3) << 1);
                    a.u[vv] = *(const unsigned int*)&xlds[arow * XP + kk];
                }
                const int brow = nt * 16 + ll;
                #pragma unroll
                for (int vv = 0; vv < 8; ++vv) {
                    const int kk = ks * 32 + ((vv >> 2) << 4) + (lh << 3) + ((vv & 3) << 1);
                    bb.u[vv] = *(const unsigned int*)&wlds[brow * WP + kk];
                }
                acc = __builtin_amdgcn_wmma_f32_16x16x32_bf16(
                    false, a.v, false, bb.v, (short)0, acc, false, false);
            }
            #pragma unroll
            for (int r = 0; r < 8; ++r) {
                const int row = blk * 64 + wid * 16 + (lh << 3) + r;
                const int o = pass * 96 + nt * 16 + ll;
                Y[(size_t)row * EMBED + o] = acc[r];
            }
        }
    }
}

// ================================================================
extern "C" void kernel_launch(void* const* d_in, const int* in_sizes, int n_in,
                              void* d_out, int out_size, void* d_ws, size_t ws_size,
                              hipStream_t stream) {
    const float* q    = (const float*)d_in[0];
    const float* k    = (const float*)d_in[1];
    const float* v    = (const float*)d_in[2];
    const float* mask = (const float*)d_in[3];
    const float* wq   = (const float*)d_in[4];
    const float* bq   = (const float*)d_in[5];
    const float* wk   = (const float*)d_in[6];
    const float* bk   = (const float*)d_in[7];
    const float* wv   = (const float*)d_in[8];
    const float* bv   = (const float*)d_in[9];
    const float* bias = (const float*)d_in[10];
    const float* pw   = (const float*)d_in[11];
    float* out        = (float*)d_out;

    unsigned short* ws = (unsigned short*)d_ws;
    const size_t QSZ = (size_t)NWIN * NHEAD * NP * KDIM;   // 50,331,648 bf16 elems
    unsigned short* oq  = ws;
    unsigned short* okk = ws + QSZ;
    unsigned short* ov  = ws + 2 * QSZ;
    unsigned short* x2  = ws + 3 * QSZ;                    // (nw, l, embed) bf16

    k_ln_pack<<<dim3(NWIN, 3), 256, 0, stream>>>(q, k, v, wq, bq, wk, bk, wv, bv,
                                                 oq, okk, ov);
    k_attn<<<NWIN * NHEAD, 128, 0, stream>>>(oq, okk, ov, bias, mask, x2);
    k_proj<<<(NWIN * LTOK) / 64, 128, 0, stream>>>(x2, pw, out);
}
